// FeatureFusionBlock_2267742732571
// MI455X (gfx1250) — compile-verified, matrix operand order checked by reading, not run on hardware
//
#include <hip/hip_runtime.h>
#include <hip/hip_bf16.h>

typedef __bf16 bf16;
typedef __attribute__((ext_vector_type(16))) __bf16 v16bf;
typedef __attribute__((ext_vector_type(8)))  __bf16 v8bf;
typedef __attribute__((ext_vector_type(8)))  float   v8f;

// ---------------------------------------------------------------------------
// Fragment load: 16-bit A/B operand layout for v_wmma_*_16x16x32_bf16.
// Lane l<16 holds K elements {k0+0..7, k0+16..23} of row (m0+l);
// lane l>=16 holds K elements {k0+8..15, k0+24..31}.  Two 16-byte loads.
// ---------------------------------------------------------------------------
static __device__ __forceinline__ v16bf load_frag(const bf16* p, int k0, int half) {
  union { v16bf v; v8bf h[2]; } u;
  u.h[0] = *(const v8bf*)(p + k0 + half * 8);
  u.h[1] = *(const v8bf*)(p + k0 + 16 + half * 8);
  return u.v;
}

// ---------------------------------------------------------------------------
// Generic batched bf16 GEMM, fp32 accumulate via v_wmma_f32_16x16x32_bf16.
//   C[p]  = A[p] (MxK, row-major, leading dim lda)
//         @ B[p] (KxN) supplied as Bt (NxK row-major, leading dim ldb)
// batch offset for X = bb*sXb + h*sXh  with p = bb*HH + h.
// Epilogue: optional accumulate into Cf, bias[col], addsrc (same strides as C),
// relu; stores to fp32 (Cf) and/or bf16 (Cb).
// Wave tile: 32x64 (8 WMMA accs, two A-frags share four B-frags).
// Block: 256 thr = 8 waves = 256x64 tile.  All M used here are %32 == 0.
// ---------------------------------------------------------------------------
__global__ __launch_bounds__(256) void gemm_wmma_bf16(
    const bf16* __restrict__ A, int lda, long long sAh, long long sAb,
    const bf16* __restrict__ Bt, int ldb, long long sBh, long long sBb,
    const float* __restrict__ bias, const float* __restrict__ addsrc,
    float* __restrict__ Cf, bf16* __restrict__ Cb,
    int ldc, long long sCh, long long sCb,
    int M, int N, int K, int mBlocks, int HH, int accum, int relu)
{
  const int wave = threadIdx.x >> 5;
  const int lane = threadIdx.x & 31;
  const int half = lane >> 4;
  const int lr   = lane & 15;
  const int mb   = blockIdx.x % mBlocks;
  const int nb   = blockIdx.x / mBlocks;
  const int row0 = mb * 256 + wave * 32;
  const int col0 = nb * 64;
  if (row0 >= M) return;
  const int p  = blockIdx.y;
  const int bb = p / HH, hh = p % HH;

  const bf16* Ar0 = A  + bb * sAb + hh * sAh + (long long)(row0 + lr) * lda;
  const bf16* Ar1 = Ar0 + (long long)16 * lda;
  const bf16* Br  = Bt + bb * sBb + hh * sBh + (long long)(col0 + lr) * ldb;

  v8f acc[8];
#pragma unroll
  for (int s = 0; s < 8; ++s) acc[s] = v8f{0.f,0.f,0.f,0.f,0.f,0.f,0.f,0.f};

  for (int k0 = 0; k0 < K; k0 += 32) {
    v16bf a0 = load_frag(Ar0, k0, half);
    v16bf a1 = load_frag(Ar1, k0, half);
#pragma unroll
    for (int s = 0; s < 4; ++s) {
      v16bf b = load_frag(Br + (long long)s * 16 * ldb, k0, half);
      acc[s]     = __builtin_amdgcn_wmma_f32_16x16x32_bf16(
          false, a0, false, b, (short)0, acc[s],     false, false);
      acc[s + 4] = __builtin_amdgcn_wmma_f32_16x16x32_bf16(
          false, a1, false, b, (short)0, acc[s + 4], false, false);
    }
  }

  const long long cbase = bb * sCb + hh * sCh;
#pragma unroll
  for (int g = 0; g < 2; ++g) {
#pragma unroll
    for (int s = 0; s < 4; ++s) {
      const int col = col0 + s * 16 + lr;
#pragma unroll
      for (int r = 0; r < 8; ++r) {
        const int row = row0 + g * 16 + half * 8 + r;  // C layout: vgpr r -> M = r + half*8
        const long long idx = cbase + (long long)row * ldc + col;
        float v = acc[g * 4 + s][r];
        if (accum)  v += Cf[idx];
        if (bias)   v += bias[col];
        if (addsrc) v += addsrc[idx];
        if (relu)   v = fmaxf(v, 0.0f);
        if (Cf) Cf[idx] = v;
        if (Cb) Cb[idx] = (bf16)v;
      }
    }
  }
}

// fp32 -> bf16 convert
__global__ __launch_bounds__(256) void cvt_f32_bf16(const float* __restrict__ in,
                                                    bf16* __restrict__ out, long long n) {
  long long i = (long long)blockIdx.x * 256 + threadIdx.x;
  if (i < n) out[i] = (bf16)in[i];
}

// batched (blockIdx.y) transpose RxC fp32 -> CxR bf16
__global__ __launch_bounds__(256) void transpose_cvt(const float* __restrict__ in,
                                                     bf16* __restrict__ out, int R, int C) {
  const long long n = (long long)R * C;
  long long i = (long long)blockIdx.x * 256 + threadIdx.x;
  const float* ip = in + (long long)blockIdx.y * n;
  bf16*        op = out + (long long)blockIdx.y * n;
  if (i < n) {
    long long r = i / C, c = i % C;
    op[c * (long long)R + r] = (bf16)ip[i];
  }
}

// mr1 = cw1[0]*mq1 + cw1[1]*mk2 + cb1 ; mr2 = cw2[0]*mq2 + cw2[1]*mk1 + cb2
// outputs: mr1 (bf16, flat), gsum = mr1+mr2 (bf16, flat),
//          mr2^T per (b,h): flat i = (p*64+d)*1024+ee -> mr2t[(p*1024+ee)*64+d]
__global__ __launch_bounds__(256) void mr_kernel(
    const float* __restrict__ mq1, const float* __restrict__ mk2,
    const float* __restrict__ mq2, const float* __restrict__ mk1,
    const float* __restrict__ cw1, const float* __restrict__ cb1,
    const float* __restrict__ cw2, const float* __restrict__ cb2,
    bf16* __restrict__ mr1b, bf16* __restrict__ gsumb, bf16* __restrict__ mr2tb)
{
  long long i = (long long)blockIdx.x * 256 + threadIdx.x;
  float r1 = cw1[0] * mq1[i] + cw1[1] * mk2[i] + cb1[0];
  float r2 = cw2[0] * mq2[i] + cw2[1] * mk1[i] + cb2[0];
  mr1b[i]  = (bf16)r1;
  gsumb[i] = (bf16)(r1 + r2);
  int ee = (int)(i & 1023);
  long long pd = i >> 10;
  int d = (int)(pd & 63);
  long long pp = pd >> 6;
  mr2tb[((pp << 10) + ee) * 64 + d] = (bf16)r2;
}

// one block per logits row: softmax((s)*0.125) -> bf16 probabilities
__global__ __launch_bounds__(256) void softmax_row(const float* __restrict__ S,
                                                   bf16* __restrict__ att, int W) {
  __shared__ float red[256];
  const long long row = blockIdx.x;
  const float* s = S + row * (long long)W;
  const int t = threadIdx.x;
  float m = -3.4e38f;
  for (int j = t; j < W; j += 256) m = fmaxf(m, s[j]);
  red[t] = m; __syncthreads();
  for (int o = 128; o > 0; o >>= 1) { if (t < o) red[t] = fmaxf(red[t], red[t + o]); __syncthreads(); }
  m = red[0]; __syncthreads();
  float sum = 0.f;
  for (int j = t; j < W; j += 256) sum += __expf((s[j] - m) * 0.125f);
  red[t] = sum; __syncthreads();
  for (int o = 128; o > 0; o >>= 1) { if (t < o) red[t] += red[t + o]; __syncthreads(); }
  const float inv = 1.0f / red[0];
  for (int j = t; j < W; j += 256)
    att[row * (long long)W + j] = (bf16)(__expf((s[j] - m) * 0.125f) * inv);
}

// out[bb,i,j] = mean over H of att[(bb*H+h)*EE + i*W + j]
__global__ __launch_bounds__(256) void mean_heads(const bf16* __restrict__ att,
                                                  float* __restrict__ out,
                                                  long long EE, int HH) {
  long long i = (long long)blockIdx.x * 256 + threadIdx.x;
  long long bb = i / EE, r = i % EE;
  float s = 0.f;
  for (int h = 0; h < HH; ++h) s += (float)att[(bb * HH + h) * EE + r];
  out[i] = s * (1.0f / HH);
}

// ---------------------------------------------------------------------------
extern "C" void kernel_launch(void* const* d_in, const int* in_sizes, int n_in,
                              void* d_out, int out_size, void* d_ws, size_t ws_size,
                              hipStream_t stream)
{
  (void)in_sizes; (void)n_in; (void)out_size; (void)ws_size;
  const float* x1 =(const float*)d_in[0];   const float* x2 =(const float*)d_in[1];
  const float* Wk1=(const float*)d_in[2];   const float* bk1=(const float*)d_in[3];
  const float* Wq1=(const float*)d_in[4];   const float* bq1=(const float*)d_in[5];
  const float* Wv1=(const float*)d_in[6];   const float* bv1=(const float*)d_in[7];
  const float* Wk2=(const float*)d_in[8];   const float* bk2=(const float*)d_in[9];
  const float* Wq2=(const float*)d_in[10];  const float* bq2=(const float*)d_in[11];
  const float* Wv2=(const float*)d_in[12];  const float* bv2=(const float*)d_in[13];
  const float* Wp1=(const float*)d_in[14];  const float* bp1=(const float*)d_in[15];
  const float* Wp2=(const float*)d_in[16];  const float* bp2=(const float*)d_in[17];
  const float* cw1=(const float*)d_in[18];  const float* cb1=(const float*)d_in[19];
  const float* cw2=(const float*)d_in[20];  const float* cb2=(const float*)d_in[21];
  const float* Wg1=(const float*)d_in[22];  const float* bg1=(const float*)d_in[23];
  const float* Wg2=(const float*)d_in[24];  const float* bg2=(const float*)d_in[25];
  const float* MK =(const float*)d_in[26];
  const float* MQ =(const float*)d_in[27];
  const float* MV =(const float*)d_in[28];

  const int Bc = 4, Ec = 1024, Nc = 768, Hc = 12;
  const long long F  = (long long)Bc * Ec * Nc;   // 3,145,728
  const long long EE = (long long)Ec * Ec;        // 1,048,576
  const int P = Bc * Hc;                          // 48
  const int Mrows = Bc * Ec;                      // 4096
  const long long s64k = 65536;                   // E*hd per head-block

  // ---- bump allocator on d_ws ----
  char* wsb = (char*)d_ws; size_t off = 0;
  auto alloc = [&](size_t bytes) -> char* {
    off = (off + 255) & ~(size_t)255;
    char* pp = wsb + off; off += bytes; return pp;
  };
  const size_t W2 = (size_t)Nc * Nc;
  bf16* Wk1t=(bf16*)alloc(W2*2); bf16* Wq1t=(bf16*)alloc(W2*2); bf16* Wv1t=(bf16*)alloc(W2*2);
  bf16* Wk2t=(bf16*)alloc(W2*2); bf16* Wq2t=(bf16*)alloc(W2*2); bf16* Wv2t=(bf16*)alloc(W2*2);
  bf16* Wp1t=(bf16*)alloc(W2*2); bf16* Wp2t=(bf16*)alloc(W2*2);
  bf16* MKt =(bf16*)alloc(W2*2); bf16* MQt =(bf16*)alloc(W2*2); bf16* MVt =(bf16*)alloc(W2*2);
  bf16* Wg1t=(bf16*)alloc((size_t)Nc*2*Nc*2);
  bf16* Wg2t=(bf16*)alloc((size_t)Nc*2*Nc*2);
  bf16* x1b=(bf16*)alloc(F*2);  bf16* x2b=(bf16*)alloc(F*2);
  bf16* k1b=(bf16*)alloc(F*2);  bf16* q1b=(bf16*)alloc(F*2);  bf16* v1b=(bf16*)alloc(F*2);
  bf16* k2b=(bf16*)alloc(F*2);  bf16* q2b=(bf16*)alloc(F*2);  bf16* v2b=(bf16*)alloc(F*2);
  float* mk1=(float*)alloc(F*4); float* mk2=(float*)alloc(F*4);
  float* mq1=(float*)alloc(F*4); float* mq2=(float*)alloc(F*4);
  bf16* mv1mb=(bf16*)alloc(F*2); bf16* mv2mb=(bf16*)alloc(F*2);
  bf16* mr1b=(bf16*)alloc(F*2);  bf16* mr2tb=(bf16*)alloc(F*2); bf16* gsumb=(bf16*)alloc(F*2);
  bf16* g1b=(bf16*)alloc((size_t)Mrows*2*Nc*2);
  float* mwf=(float*)alloc(F*4); bf16* mwTb=(bf16*)alloc(F*2);
  float* mv1f=(float*)alloc((size_t)P*64*64*4); float* mv2f=(float*)alloc((size_t)P*64*64*4);
  bf16* mv1tb=(bf16*)alloc((size_t)P*64*64*2);  bf16* mv2tb=(bf16*)alloc((size_t)P*64*64*2);
  bf16* u1tb=(bf16*)alloc(F*2);  bf16* u2tb=(bf16*)alloc(F*2);
  bf16* y1b=(bf16*)alloc(F*2);   bf16* y2b=(bf16*)alloc(F*2);
  float* S   =(float*)alloc((size_t)P*EE*4);   // logits (reused for both streams)
  bf16*  attb=(bf16*)alloc((size_t)P*EE*2);    // probabilities (reused)

  float* out = (float*)d_out;
  float* x1o = out;
  float* x2o = out + F;
  float* am1 = out + 2 * F;
  float* am2 = am1 + (long long)Bc * EE;

  auto gemm = [&](const bf16* Ap,int lda,long long sAh,long long sAb,
                  const bf16* Bp,int ldb,long long sBh,long long sBb,
                  const float* bias,const float* addsrc,
                  float* Cf, bf16* Cb, int ldc, long long sCh, long long sCb,
                  int Mm,int Nn,int Kk,int batch,int accum,int relu){
    int mBlocks = (Mm + 255) / 256;
    dim3 grid(mBlocks * (Nn / 64), batch);
    gemm_wmma_bf16<<<grid, 256, 0, stream>>>(Ap,lda,sAh,sAb, Bp,ldb,sBh,sBb,
        bias,addsrc, Cf,Cb, ldc,sCh,sCb, Mm,Nn,Kk,mBlocks,Hc,accum,relu);
  };
  auto tw = [&](const float* Win, bf16* Wo, int R, int C){
    transpose_cvt<<<dim3((unsigned)(((long long)R*C + 255)/256), 1), 256, 0, stream>>>(Win, Wo, R, C);
  };

  // ---- stage 0: convert inputs / transpose+convert weights to bf16 ----
  cvt_f32_bf16<<<dim3((unsigned)((F+255)/256)), 256, 0, stream>>>(x1, x1b, F);
  cvt_f32_bf16<<<dim3((unsigned)((F+255)/256)), 256, 0, stream>>>(x2, x2b, F);
  tw(Wk1,Wk1t,Nc,Nc); tw(Wq1,Wq1t,Nc,Nc); tw(Wv1,Wv1t,Nc,Nc);
  tw(Wk2,Wk2t,Nc,Nc); tw(Wq2,Wq2t,Nc,Nc); tw(Wv2,Wv2t,Nc,Nc);
  tw(Wp1,Wp1t,Nc,Nc); tw(Wp2,Wp2t,Nc,Nc);
  tw(MK,MKt,Nc,Nc);   tw(MQ,MQt,Nc,Nc);   tw(MV,MVt,Nc,Nc);
  tw(Wg1,Wg1t,Nc,2*Nc); tw(Wg2,Wg2t,2*Nc,Nc);

  // ---- stage 1: six projections (bias, bf16 out) ----
  gemm(x1b,Nc,0,0, Wk1t,Nc,0,0, bk1,nullptr, nullptr,k1b, Nc,0,0, Mrows,Nc,Nc, 1,0,0);
  gemm(x1b,Nc,0,0, Wq1t,Nc,0,0, bq1,nullptr, nullptr,q1b, Nc,0,0, Mrows,Nc,Nc, 1,0,0);
  gemm(x1b,Nc,0,0, Wv1t,Nc,0,0, bv1,nullptr, nullptr,v1b, Nc,0,0, Mrows,Nc,Nc, 1,0,0);
  gemm(x2b,Nc,0,0, Wk2t,Nc,0,0, bk2,nullptr, nullptr,k2b, Nc,0,0, Mrows,Nc,Nc, 1,0,0);
  gemm(x2b,Nc,0,0, Wq2t,Nc,0,0, bq2,nullptr, nullptr,q2b, Nc,0,0, Mrows,Nc,Nc, 1,0,0);
  gemm(x2b,Nc,0,0, Wv2t,Nc,0,0, bv2,nullptr, nullptr,v2b, Nc,0,0, Mrows,Nc,Nc, 1,0,0);

  // ---- stage 2: memory-path matmuls ----
  gemm(k1b,Nc,0,0, MKt,Nc,0,0, nullptr,nullptr, mk1,nullptr, Nc,0,0, Mrows,Nc,Nc, 1,0,0);
  gemm(k2b,Nc,0,0, MKt,Nc,0,0, nullptr,nullptr, mk2,nullptr, Nc,0,0, Mrows,Nc,Nc, 1,0,0);
  gemm(q1b,Nc,0,0, MQt,Nc,0,0, nullptr,nullptr, mq1,nullptr, Nc,0,0, Mrows,Nc,Nc, 1,0,0);
  gemm(q2b,Nc,0,0, MQt,Nc,0,0, nullptr,nullptr, mq2,nullptr, Nc,0,0, Mrows,Nc,Nc, 1,0,0);
  gemm(v1b,Nc,0,0, MVt,Nc,0,0, nullptr,nullptr, nullptr,mv1mb, Nc,0,0, Mrows,Nc,Nc, 1,0,0);
  gemm(v2b,Nc,0,0, MVt,Nc,0,0, nullptr,nullptr, nullptr,mv2mb, Nc,0,0, Mrows,Nc,Nc, 1,0,0);

  // ---- stage 3: mr1/mr2 elementwise + mr2 transpose ----
  mr_kernel<<<dim3((unsigned)(F/256)), 256, 0, stream>>>(mq1,mk2,mq2,mk1, cw1,cb1,cw2,cb2,
                                                          mr1b,gsumb,mr2tb);

  // ---- stage 4: gate MLP  mw = relu(relu(gsum@Wg1+bg1)@Wg2+bg2) ----
  gemm(gsumb,Nc,0,0, Wg1t,Nc,0,0, bg1,nullptr, nullptr,g1b, 2*Nc,0,0, Mrows,2*Nc,Nc, 1,0,1);
  gemm(g1b,2*Nc,0,0, Wg2t,2*Nc,0,0, bg2,nullptr, mwf,nullptr, Nc,0,0, Mrows,Nc,2*Nc, 1,0,1);
  // mw reinterpreted (P,E,64) -> mwT (P,64,E) bf16
  transpose_cvt<<<dim3(256, P), 256, 0, stream>>>(mwf, mwTb, Ec, 64);

  // ---- stage 5: mv1/mv2 = (mv?m viewed P x 64 x E) @ mw_r  -> 64x64 per head ----
  gemm(mv1mb,Ec, s64k, Hc*s64k, mwTb,Ec, s64k, Hc*s64k, nullptr,nullptr,
       mv1f,nullptr, 64, 4096, (long long)Hc*4096, 64,64,Ec, P,0,0);
  gemm(mv2mb,Ec, s64k, Hc*s64k, mwTb,Ec, s64k, Hc*s64k, nullptr,nullptr,
       mv2f,nullptr, 64, 4096, (long long)Hc*4096, 64,64,Ec, P,0,0);
  transpose_cvt<<<dim3(16, P), 256, 0, stream>>>(mv1f, mv1tb, 64, 64);
  transpose_cvt<<<dim3(16, P), 256, 0, stream>>>(mv2f, mv2tb, 64, 64);

  // ---- stage 6: u?t = (v?h @ mv?)^T = mv?^T @ v?h^T   (P x 64 x E, bf16) ----
  gemm(mv1tb,64, 4096, (long long)Hc*4096, v2b,Nc, 64, (long long)Ec*Nc, nullptr,nullptr,
       nullptr,u1tb, Ec, s64k, Hc*s64k, 64,Ec,64, P,0,0);
  gemm(mv2tb,64, 4096, (long long)Hc*4096, v1b,Nc, 64, (long long)Ec*Nc, nullptr,nullptr,
       nullptr,u2tb, Ec, s64k, Hc*s64k, 64,Ec,64, P,0,0);

  // ---- stream 1 attention: S = q1h@k2h^T ; S += mD ; softmax ; mean ; y1 ----
  gemm(q1b,Nc, 64, (long long)Ec*Nc, k2b,Nc, 64, (long long)Ec*Nc, nullptr,nullptr,
       S,nullptr, Ec, EE, Hc*EE, Ec,Ec,64, P,0,0);
  gemm(mr1b,64, s64k, Hc*s64k, mr2tb,64, s64k, Hc*s64k, nullptr,nullptr,
       S,nullptr, Ec, EE, Hc*EE, Ec,Ec,64, P,1,0);
  softmax_row<<<dim3((unsigned)(P*Ec)), 256, 0, stream>>>(S, attb, Ec);
  mean_heads<<<dim3((unsigned)((Bc*EE)/256)), 256, 0, stream>>>(attb, am1, EE, Hc);
  gemm(attb,Ec, EE, Hc*EE, u1tb,Ec, s64k, Hc*s64k, nullptr,nullptr,
       nullptr,y1b, Nc, 64, (long long)Ec*Nc, Ec,64,Ec, P,0,0);

  // ---- stream 2 attention (reuses S and attb) ----
  gemm(q2b,Nc, 64, (long long)Ec*Nc, k1b,Nc, 64, (long long)Ec*Nc, nullptr,nullptr,
       S,nullptr, Ec, EE, Hc*EE, Ec,Ec,64, P,0,0);
  gemm(mr1b,64, s64k, Hc*s64k, mr2tb,64, s64k, Hc*s64k, nullptr,nullptr,
       S,nullptr, Ec, EE, Hc*EE, Ec,Ec,64, P,1,0);
  softmax_row<<<dim3((unsigned)(P*Ec)), 256, 0, stream>>>(S, attb, Ec);
  mean_heads<<<dim3((unsigned)((Bc*EE)/256)), 256, 0, stream>>>(attb, am2, EE, Hc);
  gemm(attb,Ec, EE, Hc*EE, u2tb,Ec, s64k, Hc*s64k, nullptr,nullptr,
       nullptr,y2b, Nc, 64, (long long)Ec*Nc, Ec,64,Ec, P,0,0);

  // ---- final projections with residual add, fp32 straight to d_out ----
  gemm(y1b,Nc,0,0, Wp1t,Nc,0,0, bp1, x1, x1o,nullptr, Nc,0,0, Mrows,Nc,Nc, 1,0,0);
  gemm(y2b,Nc,0,0, Wp2t,Nc,0,0, bp2, x2, x2o,nullptr, Nc,0,0, Mrows,Nc,Nc, 1,0,0);
}
